// LiquidSpikeFormer_41051297415486
// MI455X (gfx1250) — matrix operand order repaired and checked by reading, not compiled, not run on hardware
//
#include <hip/hip_runtime.h>
#include <math.h>

// ---------------- problem constants (from setup_inputs) ----------------
#define B_   32
#define T_   64
#define P_   4096
#define E_   256
#define NH_  8
#define HD_  32
#define C_   10
#define BT_  (B_ * T_)          // 2048
#define TP_  (T_ + 2)           // time-padded rows per batch for the conv GEMM
#define SPH_ROWS_ (B_ * TP_)    // 2112 padded rows

typedef __attribute__((ext_vector_type(16))) _Float16 v16h;
typedef __attribute__((ext_vector_type(8)))  _Float16 v8h;
typedef __attribute__((ext_vector_type(8)))  float    v8f;

// ---------------- WMMA f16 fragment loaders (pre-staged f16 operands) ----------------
// A matrix 16x32 f16 layout (CDNA5 ISA 7.12.2):
//  lanes 0-15  (M=lane)   : halves 0..7 -> K=k0..k0+7,    halves 8..15 -> K=k0+16..k0+23
//  lanes 16-31 (M=lane-16): halves 0..7 -> K=k0+8..k0+15, halves 8..15 -> K=k0+24..k0+31
__device__ __forceinline__ v16h load_fragA_h(const _Float16* __restrict__ row, int k0, int hi) {
    const v8h lo = *reinterpret_cast<const v8h*>(row + k0 + hi * 8);
    const v8h hh = *reinterpret_cast<const v8h*>(row + k0 + 16 + hi * 8);
    v16h r;
    #pragma unroll
    for (int i = 0; i < 8; ++i) { r[i] = lo[i]; r[8 + i] = hh[i]; }
    return r;
}

// B matrix 32x16 f16 layout: lane = column N; lanes 0-15: halves 0..7 -> K=k0..k0+7, halves 8..15 -> K=k0+16..k0+23,
// lanes 16-31: halves 0..7 -> K=k0+8..k0+15, halves 8..15 -> K=k0+24..k0+31 (same as A).
__device__ __forceinline__ v16h load_fragB_h(const _Float16* __restrict__ row, int k0, int hi) {
    const _Float16* base = row + k0 + hi * 8;
    const v8h lo = *reinterpret_cast<const v8h*>(base);
    const v8h hh = *reinterpret_cast<const v8h*>(base + 16);
    v16h r;
    #pragma unroll
    for (int i = 0; i < 8; ++i) { r[i] = lo[i]; r[8 + i] = hh[i]; }
    return r;
}

// ---------------- generic WMMA linear: out[M,N] = act(A[M,K] . W[N,K]^T + bias) --------
// 32x32 macro-tile per wave (2x2 WMMA tiles): each fragment load amortized over 2 WMMAs.
// ntaps>1 implements the 3-tap temporal conv. A is then in a time-padded layout:
// logical row m = b*padT + t lives at padded row b*(padT+2) + 1 + t, with zeroed guard
// rows at both ends of each batch, so tap shifts are always in-bounds (no masking; the
// inner loop is pure b128-load -> WMMA and EXEC stays all-ones throughout).
__global__ void wmma_linear_kernel(
    const _Float16* __restrict__ A, int lda,
    const _Float16* __restrict__ W, int ldw, long long wTapStride,
    const float* __restrict__ bias,
    float* __restrict__ outF, _Float16* __restrict__ outH, int ldo,
    int M, int N, int K, int ntaps, int shift0, int padT, int relu)
{
    const int wave = threadIdx.x >> 5;
    const int lane = threadIdx.x & 31;
    const int tilesN = N >> 5;               // 32-wide macro tiles
    const int tilesM = M >> 5;
    const int tile = blockIdx.x * (blockDim.x >> 5) + wave;
    if (tile >= tilesM * tilesN) return;     // uniform per wave
    const int tm = tile / tilesN;
    const int tn = tile - tm * tilesN;
    const int lh = lane & 15;
    const int hi = lane >> 4;
    const int m0 = tm * 32;
    const int n0 = tn * 32;
    int srcRow0 = m0 + lh;                   // A rows this lane loads (frag 0 / frag 1)
    int srcRow1 = m0 + 16 + lh;
    if (padT > 0) {                          // padded-layout remap (conv GEMM only)
        srcRow0 += 2 * (srcRow0 / padT) + 1;
        srcRow1 += 2 * (srcRow1 / padT) + 1;
    }
    const _Float16* Wrow0base = W + (long long)(n0 + lh) * ldw;
    const _Float16* Wrow1base = W + (long long)(n0 + 16 + lh) * ldw;

    v8f acc00 = {}, acc01 = {}, acc10 = {}, acc11 = {};

    for (int tap = 0; tap < ntaps; ++tap) {
        const int shift = shift0 + tap;
        const _Float16* Arow0 = A + (long long)(srcRow0 + shift) * lda;
        const _Float16* Arow1 = A + (long long)(srcRow1 + shift) * lda;
        const _Float16* Wrow0 = Wrow0base + (long long)tap * wTapStride;
        const _Float16* Wrow1 = Wrow1base + (long long)tap * wTapStride;
        for (int k0 = 0; k0 < K; k0 += 32) {
            const v16h a0 = load_fragA_h(Arow0, k0, hi);
            const v16h a1 = load_fragA_h(Arow1, k0, hi);
            const v16h b0 = load_fragB_h(Wrow0, k0, hi);
            acc00 = __builtin_amdgcn_wmma_f32_16x16x32_f16(false, a0, false, b0, (short)0, acc00, false, false);
            acc10 = __builtin_amdgcn_wmma_f32_16x16x32_f16(false, a1, false, b0, (short)0, acc10, false, false);
            const v16h b1 = load_fragB_h(Wrow1, k0, hi);
            acc01 = __builtin_amdgcn_wmma_f32_16x16x32_f16(false, a0, false, b1, (short)0, acc01, false, false);
            acc11 = __builtin_amdgcn_wmma_f32_16x16x32_f16(false, a1, false, b1, (short)0, acc11, false, false);
        }
    }

    // D layout: lane (0-15): N=lane, VGPR v -> M=v ; lanes 16-31: N=lane-16, M=v+8
    const float bn0 = bias[n0 + lh];
    const float bn1 = bias[n0 + 16 + lh];
    #pragma unroll
    for (int v = 0; v < 8; ++v) {
        const int mA = m0 + v + hi * 8;        // rows of acc0* / acc1* tiles
        const int mB = m0 + 16 + v + hi * 8;
        float v00 = acc00[v] + bn0;
        float v01 = acc01[v] + bn1;
        float v10 = acc10[v] + bn0;
        float v11 = acc11[v] + bn1;
        if (relu) {
            v00 = fmaxf(v00, 0.0f); v01 = fmaxf(v01, 0.0f);
            v10 = fmaxf(v10, 0.0f); v11 = fmaxf(v11, 0.0f);
        }
        const long long rA = (long long)mA * ldo;
        const long long rB = (long long)mB * ldo;
        if (outF) {
            outF[rA + n0 + lh]      = v00;
            outF[rA + n0 + 16 + lh] = v01;
            outF[rB + n0 + lh]      = v10;
            outF[rB + n0 + 16 + lh] = v11;
        }
        if (outH) {
            outH[rA + n0 + lh]      = (_Float16)v00;
            outH[rA + n0 + 16 + lh] = (_Float16)v01;
            outH[rB + n0 + lh]      = (_Float16)v10;
            outH[rB + n0 + 16 + lh] = (_Float16)v11;
        }
    }
}

// ---------------- helpers ----------------
__device__ __forceinline__ float block_reduce_sum_256(float v, float* sm) {
    const int tid = threadIdx.x;
    sm[tid] = v;
    __syncthreads();
    for (int off = 128; off > 0; off >>= 1) {
        if (tid < off) sm[tid] += sm[tid + off];
        __syncthreads();
    }
    const float r = sm[0];
    __syncthreads();
    return r;
}

__global__ void f32_to_f16_kernel(const float* __restrict__ src, _Float16* __restrict__ dst) {
    const long long i = (long long)blockIdx.x * blockDim.x + threadIdx.x;
    dst[i] = (_Float16)src[i];
}

// zero the 2 guard rows per batch of the padded f16 activation matrix
__global__ void zero_pad_rows_kernel(_Float16* __restrict__ sp_h) {
    const long long i = (long long)blockIdx.x * blockDim.x + threadIdx.x;  // over B*2*P
    const int b = (int)(i / (2 * P_));
    const int r = (int)((i / P_) & 1);             // 0 = top guard, 1 = bottom guard
    const int p = (int)(i % P_);
    const long long row = (long long)b * TP_ + (r ? (TP_ - 1) : 0);
    sp_h[row * P_ + p] = (_Float16)0.0f;
}

// ---------------- 1) pe_w (E,P,3) -> wpe_h[(k*E+e)*P + p] (f16) ----------------
__global__ void transpose_pew_kernel(const float* __restrict__ pe_w, _Float16* __restrict__ wpe) {
    const long long i = (long long)blockIdx.x * blockDim.x + threadIdx.x;  // over 3*E*P
    const int kk = (int)(i / ((long long)E_ * P_));
    const int r  = (int)(i - (long long)kk * E_ * P_);
    const int e  = r / P_;
    const int p  = r - e * P_;
    wpe[i] = (_Float16)pe_w[((long long)e * P_ + p) * 3 + kk];
}

// ---------------- 2) Gaussian smooth over T + pixel LayerNorm over P ----------------
__global__ void smooth_pixel_ln_kernel(const float* __restrict__ events,
                                       const float* __restrict__ pixel_g,
                                       const float* __restrict__ pixel_b,
                                       float* __restrict__ sp,
                                       float* __restrict__ rowsum,
                                       float* __restrict__ rowsq)
{
    __shared__ float sm[256];
    const int bt = blockIdx.x;
    const int b = bt / T_, t = bt - b * T_;
    float w[5];
    float wsum = 0.0f;
    #pragma unroll
    for (int k = 0; k < 5; ++k) {
        const float c = (float)k - 2.0f;
        w[k] = expf(-c * c / (2.0f * (5.0f / 6.0f) * (5.0f / 6.0f)));
        wsum += w[k];
    }
    #pragma unroll
    for (int k = 0; k < 5; ++k) w[k] /= wsum;

    float vals[P_ / 256];
    float lsum = 0.0f, lsq = 0.0f;
    #pragma unroll
    for (int i = 0; i < P_ / 256; ++i) {
        const int p = threadIdx.x + i * 256;
        float acc = 0.0f;
        #pragma unroll
        for (int k = 0; k < 5; ++k) {
            const int tt = t + k - 2;
            if (tt >= 0 && tt < T_)
                acc += w[k] * events[((long long)b * T_ + tt) * P_ + p];
        }
        vals[i] = acc;
        lsum += acc;
        lsq  += acc * acc;
    }
    const float tot  = block_reduce_sum_256(lsum, sm);
    const float tot2 = block_reduce_sum_256(lsq, sm);
    const float mean = tot / (float)P_;
    const float var  = tot2 / (float)P_ - mean * mean;
    const float rstd = rsqrtf(var + 1e-5f);

    float nsum = 0.0f, nsq = 0.0f;
    #pragma unroll
    for (int i = 0; i < P_ / 256; ++i) {
        const int p = threadIdx.x + i * 256;
        const float nv = (vals[i] - mean) * rstd * pixel_g[p] + pixel_b[p];
        sp[(long long)bt * P_ + p] = nv;
        nsum += nv;
        nsq  += nv * nv;
    }
    const float rs  = block_reduce_sum_256(nsum, sm);
    const float rs2 = block_reduce_sum_256(nsq, sm);
    if (threadIdx.x == 0) { rowsum[bt] = rs; rowsq[bt] = rs2; }
}

// ---------------- 3) per-batch global LN statistics ----------------
__global__ void global_stats_kernel(const float* __restrict__ rowsum,
                                    const float* __restrict__ rowsq,
                                    float* __restrict__ gmean, float* __restrict__ gscale)
{
    __shared__ float s1[64], s2[64];
    const int b = blockIdx.x;
    const int tid = threadIdx.x;   // 64 threads == T_
    s1[tid] = rowsum[b * T_ + tid];
    s2[tid] = rowsq[b * T_ + tid];
    __syncthreads();
    for (int off = 32; off > 0; off >>= 1) {
        if (tid < off) { s1[tid] += s1[tid + off]; s2[tid] += s2[tid + off]; }
        __syncthreads();
    }
    if (tid == 0) {
        const float inv = 1.0f / (float)((long long)T_ * P_);
        const float m = s1[0] * inv;
        const float v = s2[0] * inv - m * m;
        gmean[b] = m;
        gscale[b] = rsqrtf(v + 1e-5f);
    }
}

// ---------------- 4) apply global LN; emit f16 copy in time-padded layout -------------
__global__ void apply_global_ln_kernel(float* __restrict__ sp, _Float16* __restrict__ sp_h,
                                       const float* __restrict__ gg, const float* __restrict__ gb,
                                       const float* __restrict__ gmean, const float* __restrict__ gscale)
{
    const long long i = (long long)blockIdx.x * blockDim.x + threadIdx.x;
    const long long TPn = (long long)T_ * P_;
    const int b = (int)(i / TPn);
    const long long tp = i - (long long)b * TPn;
    const float v = (sp[i] - gmean[b]) * gscale[b] * gg[tp] + gb[tp];
    sp[i] = v;
    // padded row = b*(T+2) + 1 + t  ->  flat = b*(T+2)*P + P + tp
    sp_h[(long long)b * TP_ * P_ + P_ + tp] = (_Float16)v;
}

// ---------------- 5) sequential LIF scan (one block per batch, tau state in LDS) ------
__global__ void lif_scan_kernel(const float* __restrict__ xt,
                                const float* __restrict__ mem,
                                const float* __restrict__ tauW,   // (E, 2E); Wtt = [:, E:]
                                const float* __restrict__ thrPtr,
                                float* __restrict__ spikes,
                                float* __restrict__ vfinal)
{
    __shared__ float tau_s[E_];
    const int b = blockIdx.x;
    const int e = threadIdx.x;    // 256 == E_
    tau_s[e] = 1.0f;
    __syncthreads();
    const float thr = thrPtr[0];
    const float* wrow = tauW + (long long)e * (2 * E_) + E_;   // Wtt row e
    float v = 0.0f;
    for (int t = 0; t < T_; ++t) {
        float acc = xt[((long long)b * T_ + t) * E_ + e];
        for (int j = 0; j < E_; ++j) acc += wrow[j] * tau_s[j];
        __syncthreads();
        const float tau = 1.0f / (1.0f + expf(-acc));
        tau_s[e] = tau;
        __syncthreads();
        const float alpha = expf(-1.0f / (tau + 1e-6f));
        const float m = mem[((long long)b * T_ + t) * E_ + e];
        v = alpha * v + (1.0f - alpha) * m;
        const float s = (v >= thr) ? 1.0f : 0.0f;
        spikes[((long long)b * T_ + t) * E_ + e] = s;
        v *= (1.0f - s);
    }
    vfinal[(long long)b * E_ + e] = v;
}

// ---------------- 6) h = spikes + sinusoidal PE (f32 + f16); copy threshold -----------
__global__ void add_pe_kernel(const float* __restrict__ spikes,
                              float* __restrict__ h, _Float16* __restrict__ h16,
                              const float* __restrict__ thrPtr, float* __restrict__ thrOut)
{
    const long long i = (long long)blockIdx.x * blockDim.x + threadIdx.x;  // over BT*E
    const int e = (int)(i % E_);
    const int t = (int)((i / E_) % T_);
    const int pair = e >> 1;
    const float div = expf((float)(2 * pair) * (-logf(10000.0f) / (float)E_));
    const float ang = (float)t * div;
    const float pe = (e & 1) ? cosf(ang) : sinf(ang);
    const float v = spikes[i] + pe;
    h[i] = v;
    h16[i] = (_Float16)v;
    if (blockIdx.x == 0 && threadIdx.x == 0) thrOut[0] = thrPtr[0];
}

// ---------------- 7) attention: one block per (b, head), one thread per query t -------
__global__ void attention_kernel(const float* __restrict__ q,
                                 const float* __restrict__ k,
                                 const float* __restrict__ v,
                                 _Float16* __restrict__ o16)
{
    const int b = blockIdx.x / NH_;
    const int hID = blockIdx.x - b * NH_;
    const int t = threadIdx.x;            // 64 == T_
    const float scale = rsqrtf((float)HD_);
    const long long base = ((long long)b * T_) * E_ + hID * HD_;

    float qreg[HD_];
    #pragma unroll
    for (int d = 0; d < HD_; ++d) qreg[d] = q[base + (long long)t * E_ + d];

    float scores[T_];
    float mx = -1e30f;
    for (int s = 0; s < T_; ++s) {
        const float* krow = k + base + (long long)s * E_;
        float dot = 0.0f;
        #pragma unroll
        for (int d = 0; d < HD_; ++d) dot += qreg[d] * krow[d];
        dot *= scale;
        scores[s] = dot;
        mx = fmaxf(mx, dot);
    }
    float den = 0.0f;
    for (int s = 0; s < T_; ++s) {
        const float ev = expf(scores[s] - mx);
        scores[s] = ev;
        den += ev;
    }
    const float inv = 1.0f / den;
    float oacc[HD_];
    #pragma unroll
    for (int d = 0; d < HD_; ++d) oacc[d] = 0.0f;
    for (int s = 0; s < T_; ++s) {
        const float wgt = scores[s] * inv;
        const float* vrow = v + base + (long long)s * E_;
        #pragma unroll
        for (int d = 0; d < HD_; ++d) oacc[d] += wgt * vrow[d];
    }
    #pragma unroll
    for (int d = 0; d < HD_; ++d) o16[base + (long long)t * E_ + d] = (_Float16)oacc[d];
}

// ---------------- 8) out = LN(hin + res) over E; optional f16 copy ----------------
__global__ void residual_ln_kernel(const float* __restrict__ hin, const float* __restrict__ res,
                                   const float* __restrict__ g, const float* __restrict__ bp,
                                   float* __restrict__ out, _Float16* __restrict__ out16)
{
    __shared__ float sm[256];
    const long long row = blockIdx.x;
    const int e = threadIdx.x;   // 256 == E_
    const float val = hin[row * E_ + e] + res[row * E_ + e];
    const float mean = block_reduce_sum_256(val, sm) / (float)E_;
    const float d = val - mean;
    const float var = block_reduce_sum_256(d * d, sm) / (float)E_;
    const float rstd = rsqrtf(var + 1e-5f);
    const float r = d * rstd * g[e] + bp[e];
    out[row * E_ + e] = r;
    if (out16) out16[row * E_ + e] = (_Float16)r;
}

// ---------------- 9) mean over T + classifier head ----------------
__global__ void head_kernel(const float* __restrict__ h2,
                            const float* __restrict__ hw, const float* __restrict__ hb,
                            float* __restrict__ logits)
{
    __shared__ float mean_s[E_];
    const int b = blockIdx.x;
    const int e = threadIdx.x;   // 256 == E_
    float s = 0.0f;
    for (int t = 0; t < T_; ++t) s += h2[((long long)b * T_ + t) * E_ + e];
    mean_s[e] = s / (float)T_;
    __syncthreads();
    if (e < C_) {
        float acc = hb[e];
        for (int j = 0; j < E_; ++j) acc += hw[e * E_ + j] * mean_s[j];
        logits[b * C_ + e] = acc;
    }
}

// ---------------- host side ----------------
static inline int gemm_blocks(int M, int N) {
    const int tiles = (M / 32) * (N / 32);   // 32x32 macro tiles, 8 waves/block
    return (tiles + 7) / 8;
}

extern "C" void kernel_launch(void* const* d_in, const int* in_sizes, int n_in,
                              void* d_out, int out_size, void* d_ws, size_t ws_size,
                              hipStream_t stream) {
    const float* events  = (const float*)d_in[0];
    const float* pixel_g = (const float*)d_in[1];
    const float* pixel_b = (const float*)d_in[2];
    const float* glob_g  = (const float*)d_in[3];
    const float* glob_b  = (const float*)d_in[4];
    const float* pe_w    = (const float*)d_in[5];
    const float* pe_b    = (const float*)d_in[6];
    const float* tau_W   = (const float*)d_in[7];
    const float* tau_b   = (const float*)d_in[8];
    const float* mem_W   = (const float*)d_in[9];
    const float* mem_b   = (const float*)d_in[10];
    const float* thr     = (const float*)d_in[11];
    const float* Wq = (const float*)d_in[12]; const float* bq = (const float*)d_in[13];
    const float* Wk = (const float*)d_in[14]; const float* bk = (const float*)d_in[15];
    const float* Wv = (const float*)d_in[16]; const float* bv = (const float*)d_in[17];
    const float* Wo = (const float*)d_in[18]; const float* bo = (const float*)d_in[19];
    const float* ln1_g = (const float*)d_in[20]; const float* ln1_b = (const float*)d_in[21];
    const float* ln2_g = (const float*)d_in[22]; const float* ln2_b = (const float*)d_in[23];
    const float* W1 = (const float*)d_in[24]; const float* b1 = (const float*)d_in[25];
    const float* W2 = (const float*)d_in[26]; const float* b2 = (const float*)d_in[27];
    const float* head_W = (const float*)d_in[28]; const float* head_b = (const float*)d_in[29];

    // -------- workspace layout: f32 region, then 16B-aligned f16 region --------
    float* ws = (float*)d_ws;
    float* sp     = ws;                          // B*T*P   = 8388608
    float* membuf = sp + 8388608;                // BT*E    = 524288 each below
    float* xtbuf  = membuf + 524288;
    float* hbuf   = xtbuf  + 524288;
    float* qb     = hbuf   + 524288;
    float* kb     = qb     + 524288;
    float* vb     = kb     + 524288;
    float* h1     = vb     + 524288;
    float* ff2    = h1     + 524288;
    float* h2     = ff2    + 524288;
    float* projb  = h2     + 524288;             // Wo projection output
    float* rowsum = projb  + 524288;             // 2048
    float* rowsq  = rowsum + 2048;
    float* gmean  = rowsq  + 2048;               // 32
    float* gscale = gmean  + 32;

    _Float16* sp_h   = (_Float16*)(gscale + 32); // padded: B*(T+2)*P = 8650752
    _Float16* wpe_h  = sp_h  + (long long)SPH_ROWS_ * P_;   // 3*E*P = 3145728
    _Float16* x_h    = wpe_h + 3145728;          // BT*E = 524288
    _Float16* h_h    = x_h   + 524288;
    _Float16* h1_h   = h_h   + 524288;
    _Float16* o_h    = h1_h  + 524288;
    _Float16* ff_h   = o_h   + 524288;           // BT*4E = 2097152
    _Float16* wmem_h = ff_h  + 2097152;          // E*E   = 65536
    _Float16* wtau_h = wmem_h + 65536;           // E*2E  = 131072
    _Float16* wq_h   = wtau_h + 131072;          // E*E each
    _Float16* wk_h   = wq_h  + 65536;
    _Float16* wv_h   = wk_h  + 65536;
    _Float16* wo_h   = wv_h  + 65536;
    _Float16* w1_h   = wo_h  + 65536;            // 4E*E = 262144
    _Float16* w2_h   = w1_h  + 262144;           // E*4E = 262144

    float* out    = (float*)d_out;
    float* logits = out;                         // 320
    float* spikes = out + B_ * C_;               // 524288
    float* vfinal = spikes + (long long)BT_ * E_;  // 8192
    float* thrOut = vfinal + B_ * E_;            // 1

    // -------- one-time weight staging to f16 + guard-row zeroing --------
    transpose_pew_kernel<<<(3 * E_ * P_) / 256, 256, 0, stream>>>(pe_w, wpe_h);
    zero_pad_rows_kernel<<<(B_ * 2 * P_) / 256, 256, 0, stream>>>(sp_h);
    f32_to_f16_kernel<<<(E_ * E_) / 256, 256, 0, stream>>>(mem_W, wmem_h);
    f32_to_f16_kernel<<<(E_ * 2 * E_) / 256, 256, 0, stream>>>(tau_W, wtau_h);
    f32_to_f16_kernel<<<(E_ * E_) / 256, 256, 0, stream>>>(Wq, wq_h);
    f32_to_f16_kernel<<<(E_ * E_) / 256, 256, 0, stream>>>(Wk, wk_h);
    f32_to_f16_kernel<<<(E_ * E_) / 256, 256, 0, stream>>>(Wv, wv_h);
    f32_to_f16_kernel<<<(E_ * E_) / 256, 256, 0, stream>>>(Wo, wo_h);
    f32_to_f16_kernel<<<(4 * E_ * E_) / 256, 256, 0, stream>>>(W1, w1_h);
    f32_to_f16_kernel<<<(4 * E_ * E_) / 256, 256, 0, stream>>>(W2, w2_h);

    // -------- front end --------
    smooth_pixel_ln_kernel<<<BT_, 256, 0, stream>>>(events, pixel_g, pixel_b, sp, rowsum, rowsq);
    global_stats_kernel<<<B_, 64, 0, stream>>>(rowsum, rowsq, gmean, gscale);
    apply_global_ln_kernel<<<(B_ * T_ * P_) / 256, 256, 0, stream>>>(sp, sp_h, glob_g, glob_b, gmean, gscale);

    // temporal-conv projection (3 taps), K=4096 -> WMMA, f16-only output
    wmma_linear_kernel<<<gemm_blocks(BT_, E_), 256, 0, stream>>>(
        sp_h, P_, wpe_h, P_, (long long)E_ * P_, pe_b, nullptr, x_h, E_,
        BT_, E_, P_, /*ntaps*/3, /*shift0*/-1, /*padT*/T_, 0);
    // mem / xt_tau projections (f32 outputs feed the scan)
    wmma_linear_kernel<<<gemm_blocks(BT_, E_), 256, 0, stream>>>(
        x_h, E_, wmem_h, E_, 0, mem_b, membuf, nullptr, E_, BT_, E_, E_, 1, 0, 0, 0);
    wmma_linear_kernel<<<gemm_blocks(BT_, E_), 256, 0, stream>>>(
        x_h, E_, wtau_h, 2 * E_, 0, tau_b, xtbuf, nullptr, E_, BT_, E_, E_, 1, 0, 0, 0);

    // sequential LIF scan -> spikes, v_final (directly into d_out)
    lif_scan_kernel<<<B_, E_, 0, stream>>>(xtbuf, membuf, tau_W, thr, spikes, vfinal);
    // h = spikes + positional encoding (f32 + f16), threshold copy
    add_pe_kernel<<<(BT_ * E_) / 256, 256, 0, stream>>>(spikes, hbuf, h_h, thr, thrOut);

    // QKV projections (f32 outputs -> VALU attention)
    wmma_linear_kernel<<<gemm_blocks(BT_, E_), 256, 0, stream>>>(
        h_h, E_, wq_h, E_, 0, bq, qb, nullptr, E_, BT_, E_, E_, 1, 0, 0, 0);
    wmma_linear_kernel<<<gemm_blocks(BT_, E_), 256, 0, stream>>>(
        h_h, E_, wk_h, E_, 0, bk, kb, nullptr, E_, BT_, E_, E_, 1, 0, 0, 0);
    wmma_linear_kernel<<<gemm_blocks(BT_, E_), 256, 0, stream>>>(
        h_h, E_, wv_h, E_, 0, bv, vb, nullptr, E_, BT_, E_, E_, 1, 0, 0, 0);
    // softmax attention -> f16 context for Wo GEMM
    attention_kernel<<<B_ * NH_, T_, 0, stream>>>(qb, kb, vb, o_h);
    // output projection
    wmma_linear_kernel<<<gemm_blocks(BT_, E_), 256, 0, stream>>>(
        o_h, E_, wo_h, E_, 0, bo, projb, nullptr, E_, BT_, E_, E_, 1, 0, 0, 0);
    // h1 = LN(h + attn_out), also f16 copy for FFN
    residual_ln_kernel<<<BT_, E_, 0, stream>>>(hbuf, projb, ln1_g, ln1_b, h1, h1_h);
    // ff = relu(h1 . W1^T + b1), f16-only output (only feeds W2 GEMM)
    wmma_linear_kernel<<<gemm_blocks(BT_, 4 * E_), 256, 0, stream>>>(
        h1_h, E_, w1_h, E_, 0, b1, nullptr, ff_h, 4 * E_, BT_, 4 * E_, E_, 1, 0, 0, 1);
    // ff2 = ff . W2^T + b2
    wmma_linear_kernel<<<gemm_blocks(BT_, E_), 256, 0, stream>>>(
        ff_h, 4 * E_, w2_h, 4 * E_, 0, b2, ff2, nullptr, E_, BT_, E_, 4 * E_, 1, 0, 0, 0);
    // h2 = LN(h1 + ff2)
    residual_ln_kernel<<<BT_, E_, 0, stream>>>(h1, ff2, ln2_g, ln2_b, h2, nullptr);
    // logits = mean_T(h2) . head_W^T + head_b
    head_kernel<<<B_, E_, 0, stream>>>(h2, head_W, head_b, logits);
}
